// PointRnn_49134425866978
// MI455X (gfx1250) — matrix operand order repaired and checked by reading, not compiled, not run on hardware
//
#include <hip/hip_runtime.h>
#include <hip/hip_bf16.h>

typedef __attribute__((ext_vector_type(16))) _Float16 v16h;
typedef __attribute__((ext_vector_type(8)))  _Float16 v8h;
typedef __attribute__((ext_vector_type(8)))  float    v8f;
typedef __attribute__((ext_vector_type(4)))  float    v4f;
typedef unsigned int u32x4 __attribute__((ext_vector_type(4)));
typedef int          i32x4 __attribute__((ext_vector_type(4)));
typedef int          i32x8 __attribute__((ext_vector_type(8)));

#define BATCH 16
#define NPTS  512
#define BN    (BATCH*NPTS)   // 8192
#define KNN   8
#define SEQ   12
#define HALF  6

#if __has_builtin(__builtin_amdgcn_tensor_load_to_lds) && __has_builtin(__builtin_amdgcn_s_wait_tensorcnt)
#define HAVE_TDM 1
#else
#define HAVE_TDM 0
#endif

// ---------------------------------------------------------------------------
// Weight prep: Wt[n*Kdim + k] = (f16) W[(k+rowoff)*Nout + n]   (transpose+cvt)
// ---------------------------------------------------------------------------
__global__ void transpose_w_kernel(const float* __restrict__ W, int rowoff,
                                   int Kdim, int Nout, _Float16* __restrict__ Wt) {
    int id = blockIdx.x * 256 + threadIdx.x;
    if (id >= Kdim * Nout) return;
    int k = id / Nout;
    int n = id % Nout;
    Wt[(size_t)n * Kdim + k] = (_Float16)W[(size_t)(k + rowoff) * Nout + n];
}

__global__ void zero_kernel(float* __restrict__ p, int n) {
    int id = blockIdx.x * 256 + threadIdx.x;
    if (id < n) p[id] = 0.0f;
}

__global__ void copy_frame_kernel(const float* __restrict__ frames, int t,
                                  float* __restrict__ dst) {
    int id = blockIdx.x * 256 + threadIdx.x;
    if (id >= BATCH * NPTS * 3) return;
    int b = id / (NPTS * 3);
    int r = id % (NPTS * 3);
    dst[id] = frames[((size_t)b * SEQ + t) * NPTS * 3 + r];
}

__global__ void copy_buf_kernel(const float* __restrict__ src, float* __restrict__ dst, int n) {
    int id = blockIdx.x * 256 + threadIdx.x;
    if (id < n) dst[id] = src[id];
}

// ---------------------------------------------------------------------------
// KNN: for each query point in `cur`, top-8 nearest in `prev` (same batch).
// 2 blocks per batch, 256 threads; source cloud staged in LDS — via the
// Tensor Data Mover (1-D tile, 1536 f32) when available, issued by wave 0
// only (TDM ignores EXEC; one issue per wave), completion via TENSORcnt.
// Tie-break matches jax.lax.top_k (stable: lower index first).
// ---------------------------------------------------------------------------
__global__ __launch_bounds__(256) void knn_kernel(const float* __restrict__ cur,
                                                  const float* __restrict__ prev,
                                                  int* __restrict__ idx,
                                                  float* __restrict__ d2out) {
    __shared__ float spts[NPTS * 3];
    int b    = blockIdx.x >> 1;
    int hblk = blockIdx.x & 1;
    int tid  = threadIdx.x;
    const float* pb = prev + (size_t)b * NPTS * 3;

#if HAVE_TDM
    if (tid < 32) {
        unsigned lds_off = (unsigned)(size_t)(void*)&spts[0];
        unsigned long long ga = (unsigned long long)(size_t)pb;
        // --- D# group 0: count=1 | lds_addr | global_addr[56:0] | type=2 ---
        u32x4 g0;
        g0.x = 1u;                                              // count=1, no gather
        g0.y = lds_off;                                         // lds_addr (bytes)
        g0.z = (unsigned)(ga & 0xFFFFFFFFu);                    // global_addr[31:0]
        g0.w = (unsigned)((ga >> 32) & 0x01FFFFFFu) | (2u << 30); // addr[56:32] | type=2
        // --- D# group 1: data_size=4B, tensor_dim0=tile_dim0=1536, 1-D ---
        i32x8 g1;
        g1[0] = 2 << 16;                 // data_size=2 (4 bytes); wg_mask=0; no pad/iter
        g1[1] = (NPTS * 3) << 16;        // tensor_dim0[15:0] in bits[63:48]
        g1[2] = 1 << 16;                 // tensor_dim0[31:16]=0 | tensor_dim1=1
        g1[3] = (NPTS * 3) << 16;        // tensor_dim1 hi=0 | tile_dim0=1536
        g1[4] = 0;                       // tile_dim1=0, tile_dim2=0 (unused)
        g1[5] = NPTS * 3;                // tensor_dim0_stride[31:0]
        g1[6] = 0;
        g1[7] = 0;
        i32x4 gz = {0, 0, 0, 0};
#if __clang_major__ >= 23
        i32x8 gz8 = {0, 0, 0, 0, 0, 0, 0, 0};
        __builtin_amdgcn_tensor_load_to_lds(g0, g1, gz, gz, gz8, 0);
#else
        __builtin_amdgcn_tensor_load_to_lds(g0, g1, gz, gz, 0);
#endif
        __builtin_amdgcn_s_wait_tensorcnt(0);
    }
    __syncthreads();
#else
    for (int j = tid; j < NPTS * 3; j += 256) spts[j] = pb[j];
    __syncthreads();
#endif

    int i = hblk * 256 + tid;
    const float* q = cur + ((size_t)b * NPTS + i) * 3;
    float qx = q[0], qy = q[1], qz = q[2];
    float bd[KNN];
    int   bi[KNN];
#pragma unroll
    for (int m = 0; m < KNN; ++m) { bd[m] = 3.4e38f; bi[m] = 0; }
    for (int j = 0; j < NPTS; ++j) {
        float dx = qx - spts[j * 3 + 0];
        float dy = qy - spts[j * 3 + 1];
        float dz = qz - spts[j * 3 + 2];
        float d = dx * dx + dy * dy + dz * dz;
        if (d < bd[KNN - 1]) {
            int m = KNN - 1;
            while (m > 0 && d < bd[m - 1]) { bd[m] = bd[m - 1]; bi[m] = bi[m - 1]; --m; }
            bd[m] = d; bi[m] = j;
        }
    }
    size_t base = ((size_t)b * NPTS + i) * KNN;
#pragma unroll
    for (int m = 0; m < KNN; ++m) { idx[base + m] = bi[m]; d2out[base + m] = bd[m]; }
}

// ---------------------------------------------------------------------------
// WMMA GEMM:  C[M,Nout] = A[M,Kdim](f32, lda) @ Wt^T  (+bias) (+relu)
// Wt is (Nout x Kdim) row-major f16 (pre-transposed weights).
// One wave (32 threads) computes a 16x64 tile: A fragment reused across 4 B
// fragments; next A K-tile prefetched unconditionally (speculative
// global_prefetch_b8, WGP scope) so the loop body stays branch-free and the
// compiler can pipeline it. Kdim % 32 == 0, M % 16 == 0, Nout % 64 == 0.
// ---------------------------------------------------------------------------
__global__ __launch_bounds__(32) void gemm_wmma_kernel(
    const float* __restrict__ A, int lda,
    const _Float16* __restrict__ Wt, int ldw,
    float* __restrict__ C, const float* __restrict__ bias,
    int Kdim, int Nout, int relu) {
    int lane    = threadIdx.x;
    int halfSel = lane >> 4;    // 0: lanes 0-15, 1: lanes 16-31
    int lr      = lane & 15;
    int r0      = blockIdx.x * 16;
    int c0      = blockIdx.y * 64;

    v8f acc[4] = {};

    const float* Arow = A + (size_t)(r0 + lr) * lda;
    for (int k0 = 0; k0 < Kdim; k0 += 32) {
        // --- A fragment: 16x32 f16. lane<16 -> K {0..7,16..23}, lane>=16 -> K {8..15,24..31}
        int ka = k0 + halfSel * 8;
        v4f a0 = *(const v4f*)(Arow + ka);
        v4f a1 = *(const v4f*)(Arow + ka + 4);
        v4f a2 = *(const v4f*)(Arow + ka + 16);
        v4f a3 = *(const v4f*)(Arow + ka + 20);
        // speculative prefetch of next K tile (safe past end; keeps loop branch-free)
        __builtin_prefetch(Arow + ka + 32, 0, 3);
        v16h afrag;
#pragma unroll
        for (int e = 0; e < 4; ++e) {
            afrag[e]      = (_Float16)a0[e];
            afrag[4 + e]  = (_Float16)a1[e];
            afrag[8 + e]  = (_Float16)a2[e];
            afrag[12 + e] = (_Float16)a3[e];
        }
#pragma unroll
        for (int cc = 0; cc < 4; ++cc) {
            // --- B fragment: 32x16 f16; lane holds column (lr), lane<16 -> K {0..7,16..23}, lane>=16 -> K {8..15,24..31}
            int col = c0 + cc * 16 + lr;
            const _Float16* wp = Wt + (size_t)col * ldw + k0 + halfSel * 8;
            v8h b0 = *(const v8h*)wp;
            v8h b1 = *(const v8h*)(wp + 16);
            v16h bfrag;
#pragma unroll
            for (int e = 0; e < 8; ++e) { bfrag[e] = b0[e]; bfrag[8 + e] = b1[e]; }
            acc[cc] = __builtin_amdgcn_wmma_f32_16x16x32_f16(
                false, afrag, false, bfrag, (short)0, acc[cc], false, false);
        }
    }

#pragma unroll
    for (int cc = 0; cc < 4; ++cc) {
        int col  = c0 + cc * 16 + lr;
        float bv = bias ? bias[col] : 0.0f;
#pragma unroll
        for (int v = 0; v < 8; ++v) {
            float x = acc[cc][v] + bv;
            if (relu) x = fmaxf(x, 0.0f);
            C[(size_t)(r0 + v + halfSel * 8) * Nout + col] = x;
        }
    }
}

// ---------------------------------------------------------------------------
// Max-gather cell epilogue:
//   out[p,n] = max_k( P[ek_k, n] + disp_k . Wd[:,n] ) + (Q[p,n]?) + (bias[n]?)
// where ek_k = (d2_k <= r2) ? idx_k : idx_0, disp_k = s_xyz[ek_k] - xyz[p].
// Wd = first 3 rows of the original f32 weight matrix.
// ---------------------------------------------------------------------------
template <int COUT>
__global__ __launch_bounds__(256) void gather_max_kernel(
    const float* __restrict__ P, const float* __restrict__ Q,
    const float* __restrict__ bias, const float* __restrict__ W,
    const int* __restrict__ idx, const float* __restrict__ d2,
    const float* __restrict__ xyz, const float* __restrict__ sxyz,
    float r2, float* __restrict__ out) {
    constexpr int PPB = 256 / COUT;
    int lp = threadIdx.x / COUT;
    int n  = threadIdx.x % COUT;
    int p  = blockIdx.x * PPB + lp;
    int b  = p >> 9;

    __shared__ int   s_ek[PPB * KNN];
    __shared__ float s_dx[PPB * KNN], s_dy[PPB * KNN], s_dz[PPB * KNN];

    if (n < KNN) {
        int k = n;
        size_t base = (size_t)p * KNN;
        int e0 = idx[base];
        int ek = idx[base + k];
        float dd = d2[base + k];
        if (dd > r2) ek = e0;
        const float* sp = sxyz + ((size_t)b * NPTS + ek) * 3;
        const float* qp = xyz + (size_t)p * 3;
        s_ek[lp * KNN + k] = ek;
        s_dx[lp * KNN + k] = sp[0] - qp[0];
        s_dy[lp * KNN + k] = sp[1] - qp[1];
        s_dz[lp * KNN + k] = sp[2] - qp[2];
    }
    __syncthreads();

    float wd0 = W[n], wd1 = W[COUT + n], wd2 = W[2 * COUT + n];
    float q = bias ? bias[n] : 0.0f;
    if (Q) q += Q[(size_t)p * COUT + n];

    float best = -3.4e38f;
#pragma unroll
    for (int k = 0; k < KNN; ++k) {
        int ek  = s_ek[lp * KNN + k];
        float v = P[((size_t)b * NPTS + ek) * COUT + n]
                + s_dx[lp * KNN + k] * wd0
                + s_dy[lp * KNN + k] * wd1
                + s_dz[lp * KNN + k] * wd2;
        best = fmaxf(best, v);
    }
    out[(size_t)p * COUT + n] = best + q;
}

// ---------------------------------------------------------------------------
// Motion head: motion = h @ Wl + bl ; frame += motion ; record prediction.
// ---------------------------------------------------------------------------
__global__ __launch_bounds__(256) void motion_update_kernel(
    const float* __restrict__ h, const float* __restrict__ Wl,
    const float* __restrict__ bl, float* __restrict__ frame,
    float* __restrict__ dout, int step) {
    int p = blockIdx.x * 256 + threadIdx.x;
    if (p >= BN) return;
    const float* hp = h + (size_t)p * 64;
    float m0 = bl[0], m1 = bl[1], m2 = bl[2];
#pragma unroll 4
    for (int j = 0; j < 64; ++j) {
        float hv = hp[j];
        m0 += hv * Wl[j * 3 + 0];
        m1 += hv * Wl[j * 3 + 1];
        m2 += hv * Wl[j * 3 + 2];
    }
    float x = frame[p * 3 + 0] + m0;
    float y = frame[p * 3 + 1] + m1;
    float z = frame[p * 3 + 2] + m2;
    frame[p * 3 + 0] = x; frame[p * 3 + 1] = y; frame[p * 3 + 2] = z;
    int b = p >> 9, i = p & (NPTS - 1);
    size_t o = (((size_t)b * HALF + step) * NPTS + i) * 3;
    dout[o + 0] = x; dout[o + 1] = y; dout[o + 2] = z;
}

// ---------------------------------------------------------------------------
extern "C" void kernel_launch(void* const* d_in, const int* in_sizes, int n_in,
                              void* d_out, int out_size, void* d_ws, size_t ws_size,
                              hipStream_t stream) {
    (void)in_sizes; (void)n_in; (void)out_size; (void)ws_size;

    const float* frames = (const float*)d_in[0];
    const float* W1 = (const float*)d_in[1];  const float* b1 = (const float*)d_in[2];
    const float* W2 = (const float*)d_in[3];  const float* b2 = (const float*)d_in[4];
    const float* W3 = (const float*)d_in[5];  const float* b3 = (const float*)d_in[6];
    const float* Wm = (const float*)d_in[7];  const float* bm = (const float*)d_in[8];
    const float* Wl = (const float*)d_in[9];  const float* bl = (const float*)d_in[10];
    float* dout = (float*)d_out;

    // --- workspace carve-up (aligned to 256B) ---
    char* ws = (char*)d_ws;
    size_t off = 0;
    auto alloc = [&](size_t bytes) -> void* {
        void* p = ws + off;
        off = (off + bytes + 255) & ~(size_t)255;
        return p;
    };
    _Float16* w1n = (_Float16*)alloc(64  * 64  * 2);
    _Float16* w2f = (_Float16*)alloc(128 * 64  * 2);
    _Float16* w2n = (_Float16*)alloc(128 * 128 * 2);
    _Float16* w3f = (_Float16*)alloc(256 * 128 * 2);
    _Float16* w3n = (_Float16*)alloc(256 * 256 * 2);
    _Float16* wmt = (_Float16*)alloc(64  * 256 * 2);
    float* xyzA  = (float*)alloc(BN * 3 * 4);
    float* xyzB  = (float*)alloc(BN * 3 * 4);
    float* framebuf = (float*)alloc(BN * 3 * 4);
    int*   idxb  = (int*)  alloc(BN * KNN * 4);
    float* d2b   = (float*)alloc(BN * KNN * 4);
    float* s1    = (float*)alloc((size_t)BN * 64  * 4);
    float* s2    = (float*)alloc((size_t)BN * 128 * 4);
    float* s3    = (float*)alloc((size_t)BN * 256 * 4);
    float* Pbuf  = (float*)alloc((size_t)BN * 256 * 4);   // also reused as `h`
    float* Qbuf  = (float*)alloc((size_t)BN * 256 * 4);

    const float r2_1 = (float)((4.0 + 1e-6) * (4.0 + 1e-6));
    const float r2_2 = (float)((8.0 + 1e-6) * (8.0 + 1e-6));
    const float r2_3 = (float)((12.0 + 1e-6) * (12.0 + 1e-6));

    auto launch_transpose = [&](const float* W, int rowoff, int Kd, int No, _Float16* Wt) {
        int n = Kd * No;
        transpose_w_kernel<<<(n + 255) / 256, 256, 0, stream>>>(W, rowoff, Kd, No, Wt);
    };
    auto launch_gemm = [&](const float* A, int lda, const _Float16* Wt, int ldw,
                           float* C, const float* bias, int Kd, int No, int relu) {
        dim3 grid(BN / 16, No / 64);
        gemm_wmma_kernel<<<grid, 32, 0, stream>>>(A, lda, Wt, ldw, C, bias, Kd, No, relu);
    };
    auto launch_zero = [&](float* p, int n) {
        zero_kernel<<<(n + 255) / 256, 256, 0, stream>>>(p, n);
    };

    // --- once-per-launch weight transposes (f32 -> f16, (cout x K) row-major) ---
    launch_transpose(W1,   3, 64,  64,  w1n);
    launch_transpose(W2,   3, 64,  128, w2f);
    launch_transpose(W2,  67, 128, 128, w2n);
    launch_transpose(W3,   3, 128, 256, w3f);
    launch_transpose(W3, 131, 256, 256, w3n);
    launch_transpose(Wm,   0, 256, 64,  wmt);

    float* cur  = xyzA;
    float* prev = xyzA;   // step 0: s_xyz == xyz
    const int ncopy = BN * 3;

    for (int t = 0; t < SEQ; ++t) {
        // --- current xyz ---
        if (t < HALF) {
            copy_frame_kernel<<<(ncopy + 255) / 256, 256, 0, stream>>>(frames, t, cur);
        } else {
            if (t == HALF) {
                // input_frame = frames[:, half-1]
                copy_frame_kernel<<<(ncopy + 255) / 256, 256, 0, stream>>>(frames, HALF - 1, framebuf);
            }
            copy_buf_kernel<<<(ncopy + 255) / 256, 256, 0, stream>>>(framebuf, cur, ncopy);
        }

        // --- shared KNN for all three cells ---
        knn_kernel<<<BATCH * 2, 256, 0, stream>>>(cur, prev, idxb, d2b);

        // --- cell 1 (cout=64): P1 = s1 @ W1n ; s1 = maxgather + b1 ---
        if (t == 0) launch_zero(Pbuf, BN * 64);
        else        launch_gemm(s1, 64, w1n, 64, Pbuf, nullptr, 64, 64, 0);
        gather_max_kernel<64><<<BN / 4, 256, 0, stream>>>(
            Pbuf, nullptr, b1, W1, idxb, d2b, cur, prev, r2_1, s1);

        // --- cell 2 (cout=128): Q2 = s1 @ W2f + b2 ; P2 = s2 @ W2n ---
        launch_gemm(s1, 64, w2f, 64, Qbuf, b2, 64, 128, 0);
        if (t == 0) launch_zero(Pbuf, BN * 128);
        else        launch_gemm(s2, 128, w2n, 128, Pbuf, nullptr, 128, 128, 0);
        gather_max_kernel<128><<<BN / 2, 256, 0, stream>>>(
            Pbuf, Qbuf, nullptr, W2, idxb, d2b, cur, prev, r2_2, s2);

        // --- cell 3 (cout=256): Q3 = s2 @ W3f + b3 ; P3 = s3 @ W3n ---
        launch_gemm(s2, 128, w3f, 128, Qbuf, b3, 128, 256, 0);
        if (t == 0) launch_zero(Pbuf, BN * 256);
        else        launch_gemm(s3, 256, w3n, 256, Pbuf, nullptr, 256, 256, 0);
        gather_max_kernel<256><<<BN, 256, 0, stream>>>(
            Pbuf, Qbuf, nullptr, W3, idxb, d2b, cur, prev, r2_3, s3);

        // --- prediction head ---
        if (t >= HALF) {
            launch_gemm(s3, 256, wmt, 256, Pbuf /* h */, bm, 256, 64, 1);
            motion_update_kernel<<<(BN + 255) / 256, 256, 0, stream>>>(
                Pbuf, Wl, bl, framebuf, dout, t - HALF);
        }

        // --- ping-pong xyz buffers ---
        prev = cur;
        cur  = (prev == xyzA) ? xyzB : xyzA;
    }
}
